// CircadianEncoder_78237124264595
// MI455X (gfx1250) — matrix operand order repaired and checked by reading, not compiled, not run on hardware
//
#include <hip/hip_runtime.h>
#include <hip/hip_bf16.h>

// ---------------------------------------------------------------------------
// CircadianEncoder for MI455X (gfx1250, wave32, WMMA).
// f16 operand buffers + v_wmma_f32_16x16x32_f16 everywhere; attention is
// flash-style with paired key tiles so the PV WMMA runs at full K=32.
// ---------------------------------------------------------------------------

typedef __attribute__((ext_vector_type(16))) _Float16 v16h;
typedef __attribute__((ext_vector_type(8)))  float    v8f;
typedef __attribute__((ext_vector_type(8)))  _Float16 h8;
typedef __attribute__((ext_vector_type(4)))  _Float16 h4;

#define D_MODEL 128
#define NHEAD   8
#define HD      16
#define FF      512
#define BATCH   16
#define SEQ     1024
#define ROWS    (BATCH * SEQ)          // 16384 token rows

// ---------------------------------------------------------------------------
// fp32 -> f16 bulk convert (one float4 -> h4 per thread), n divisible by 1024
// ---------------------------------------------------------------------------
__global__ __launch_bounds__(256) void cvt_f16_kernel(
    const float* __restrict__ in, _Float16* __restrict__ out)
{
  int i = blockIdx.x * 256 + threadIdx.x;
  float4 v = ((const float4*)in)[i];
  h4 o;
  o[0] = (_Float16)v.x; o[1] = (_Float16)v.y;
  o[2] = (_Float16)v.z; o[3] = (_Float16)v.w;
  ((h4*)out)[i] = o;
}

// ---------------------------------------------------------------------------
// Embedding: x[b,s,:] = act_emb[id] + pos_enc(hours) + day_emb[dow]
// Writes fp32 x (residual/pool path) and f16 x16 (WMMA operand path).
// ---------------------------------------------------------------------------
__global__ __launch_bounds__(256) void embed_kernel(
    const int* __restrict__ act, const float* __restrict__ hours,
    const int* __restrict__ dow, const float* __restrict__ act_emb,
    const float* __restrict__ day_emb, float* __restrict__ x,
    _Float16* __restrict__ x16)
{
  int idx = blockIdx.x * 256 + threadIdx.x;       // over ROWS*64
  int t   = idx & 63;
  int row = idx >> 6;
  float h = hours[row];
  // div = 10000^(t/32)  ->  inv = exp2(-t * log2(10000)/32)
  float inv_div = exp2f(-(float)t * (13.2877123795494f / 32.0f));
  float ang = h * inv_div;
  int id = act[row];
  int dd = dow[row];
  size_t base = (size_t)row * D_MODEL;
  float v0 = act_emb[id*D_MODEL + 2*t]     + sinf(ang) + day_emb[dd*D_MODEL + 2*t];
  float v1 = act_emb[id*D_MODEL + 2*t + 1] + cosf(ang) + day_emb[dd*D_MODEL + 2*t + 1];
  x[base + 2*t]     = v0;
  x[base + 2*t + 1] = v1;
  x16[base + 2*t]     = (_Float16)v0;
  x16[base + 2*t + 1] = (_Float16)v1;
}

// ---------------------------------------------------------------------------
// WMMA GEMM:  C[M,N] = A[M,K] * W[N,K]^T + bias, optional ReLU.
// A, W already f16.  One wave per 16x16 tile, K in steps of 32.
// Fragment slices are contiguous: A = 2 x 16B loads, B = 2 x 16B loads.
// Output fp32 (Cf) or f16 (Ch) selected by non-null pointer.
// ---------------------------------------------------------------------------
__global__ __launch_bounds__(256) void gemm16_kernel(
    const _Float16* __restrict__ A, const _Float16* __restrict__ W,
    const float* __restrict__ bias, float* __restrict__ Cf,
    _Float16* __restrict__ Ch, int M, int N, int K, int relu)
{
  int wave = threadIdx.x >> 5;
  int lane = threadIdx.x & 31;
  int tilesN = N >> 4;
  int tile = blockIdx.x * 8 + wave;
  if (tile >= (M >> 4) * tilesN) return;
  int tm = tile / tilesN, tn = tile - tm * tilesN;
  int hl  = lane >> 4;            // half-lane (0/1)
  int l15 = lane & 15;
  const _Float16* Ap = A + (size_t)(tm * 16 + l15) * K;   // A row m = lane&15
  const _Float16* Wp = W + (size_t)(tn * 16 + l15) * K;   // W row = output col
  v8f acc = {};
  for (int kb = 0; kb < K; kb += 32) {
    // A 16-bit layout: elem j<8 -> k = kb + hl*8 + j ; elem j>=8 -> +16
    h8 alo = *(const h8*)(Ap + kb + hl * 8);
    h8 ahi = *(const h8*)(Ap + kb + 16 + hl * 8);
    // B 16-bit layout: elem j<8 -> k = kb + hl*8 + j ; elem j>=8 -> +16, column = lane&15
    h8 b0  = *(const h8*)(Wp + kb + hl * 8);
    h8 b1  = *(const h8*)(Wp + kb + 16 + hl * 8);
    v16h a, b;
#pragma unroll
    for (int j = 0; j < 8; ++j) {
      a[j] = alo[j]; a[j + 8] = ahi[j];
      b[j] = b0[j];  b[j + 8] = b1[j];
    }
    acc = __builtin_amdgcn_wmma_f32_16x16x32_f16(false, a, false, b,
                                                 (short)0, acc, false, false);
  }
  float bv = bias[tn * 16 + l15];
#pragma unroll
  for (int r = 0; r < 8; ++r) {                 // row m = r + 8*hl
    size_t m = (size_t)(tm * 16 + r + hl * 8);
    float v = acc[r] + bv;
    if (relu) v = fmaxf(v, 0.0f);
    if (Ch) Ch[m * N + tn * 16 + l15] = (_Float16)v;
    else    Cf[m * N + tn * 16 + l15] = v;
  }
}

// ---------------------------------------------------------------------------
// Flash attention on f16 qkv [ROWS, 384] (q|k|v, head h at h*16).
// One wave per (b, h, 16-query tile).  Keys processed 32 at a time:
// 2 padded QK^T WMMAs + joint online softmax + ONE full-K=32 PV WMMA.
// Output = f16 context [ROWS, 128] (feeds out-proj GEMM directly).
// ---------------------------------------------------------------------------
__global__ __launch_bounds__(256) void attn_kernel(
    const _Float16* __restrict__ qkv, _Float16* __restrict__ out)
{
  __shared__ float Pt[8][16][33];               // per-wave 16x32 P tile (pad)
  int wave = threadIdx.x >> 5;
  int lane = threadIdx.x & 31;
  int tile = blockIdx.x * 8 + wave;             // 16*8*64 = 8192 tiles
  int b  = tile >> 9;
  int h  = (tile >> 6) & 7;
  int qt = tile & 63;
  int hl = lane >> 4, l15 = lane & 15;

  // Q fragment (A layout, HD=16 padded to K=32: elems 8..15 zero)
  const _Float16* qp = qkv + (size_t)(b * SEQ + qt * 16 + l15) * 384 + h * HD;
  h8 qv = *(const h8*)(qp + hl * 8);
  v16h aQ;
#pragma unroll
  for (int j = 0; j < 8; ++j) { aQ[j] = qv[j]; aQ[j + 8] = (_Float16)0.0f; }

  v8f oacc = {};
  float Mrow[8], Lrow[8];
#pragma unroll
  for (int r = 0; r < 8; ++r) { Mrow[r] = -1e30f; Lrow[r] = 0.0f; }

  for (int kt = 0; kt < 32; ++kt) {             // 32 keys per iteration
    int key0 = kt * 32;
    // K fragments for the two 16-key subtiles (B layout, rows=hd, zero-pad)
    v16h bK0 = {}, bK1 = {};
    {
      const _Float16* kp0 = qkv + (size_t)(b * SEQ + key0 + l15) * 384 + 128 + h * HD + hl * 8;
      const _Float16* kp1 = kp0 + (size_t)16 * 384;
      h8 k0a = *(const h8*)(kp0);
      h8 k1a = *(const h8*)(kp1);
#pragma unroll
      for (int j = 0; j < 8; ++j) {
        bK0[j] = k0a[j];
        bK1[j] = k1a[j];
      }
    }
    v8f zero = {};
    v8f s0 = __builtin_amdgcn_wmma_f32_16x16x32_f16(false, aQ, false, bK0,
                                                    (short)0, zero, false, false);
    v8f s1 = __builtin_amdgcn_wmma_f32_16x16x32_f16(false, aQ, false, bK1,
                                                    (short)0, zero, false, false);
    // joint online softmax over the 32 keys
    float p0[8], p1[8];
#pragma unroll
    for (int r = 0; r < 8; ++r) {
      float a0 = s0[r] * 0.25f, a1 = s1[r] * 0.25f;   // 1/sqrt(HD)
      float tmax = fmaxf(a0, a1);
#pragma unroll
      for (int msk = 1; msk < 16; msk <<= 1)
        tmax = fmaxf(tmax, __shfl_xor(tmax, msk, 16));
      float newM = fmaxf(Mrow[r], tmax);
      float resc = __expf(Mrow[r] - newM);
      p0[r] = __expf(a0 - newM);
      p1[r] = __expf(a1 - newM);
      float tsum = p0[r] + p1[r];
#pragma unroll
      for (int msk = 1; msk < 16; msk <<= 1)
        tsum += __shfl_xor(tsum, msk, 16);
      Lrow[r] = Lrow[r] * resc + tsum;
      oacc[r] *= resc;
      Mrow[r] = newM;
    }
    // P: C-layout -> LDS -> A-layout (same-wave LDS is in-order; wait dscnt)
#pragma unroll
    for (int r = 0; r < 8; ++r) {
      Pt[wave][r + hl * 8][l15]      = p0[r];
      Pt[wave][r + hl * 8][16 + l15] = p1[r];
    }
    __asm__ volatile("s_wait_dscnt 0x0" ::: "memory");
    v16h aP;                                    // full K=32, no padding
#pragma unroll
    for (int j = 0; j < 8; ++j) {
      aP[j]     = (_Float16)Pt[wave][l15][hl * 16 + j];
      aP[j + 8] = (_Float16)Pt[wave][l15][hl * 16 + 8 + j];
    }
    // V fragment (B layout): elem j -> key = key0 + hl*16 + j, col hd = l15
    v16h bV;
    const _Float16* vp = qkv + (size_t)(b * SEQ + key0 + hl * 16) * 384 + 256 + h * HD + l15;
#pragma unroll
    for (int j = 0; j < 16; ++j)
      bV[j] = vp[(size_t)j * 384];
    oacc = __builtin_amdgcn_wmma_f32_16x16x32_f16(false, aP, false, bV,
                                                  (short)0, oacc, false, false);
  }
#pragma unroll
  for (int r = 0; r < 8; ++r) {
    size_t m = (size_t)(b * SEQ + qt * 16 + r + hl * 8);
    out[m * D_MODEL + h * HD + l15] = (_Float16)(oacc[r] / Lrow[r]);
  }
}

// ---------------------------------------------------------------------------
// x = LayerNorm(x + o) * g + b; writes fp32 x and f16 x16 (one wave per row)
// ---------------------------------------------------------------------------
__global__ __launch_bounds__(256) void add_ln_kernel(
    float* __restrict__ x, const float* __restrict__ o,
    const float* __restrict__ g, const float* __restrict__ bb,
    _Float16* __restrict__ x16)
{
  int wave = threadIdx.x >> 5, lane = threadIdx.x & 31;
  int row = blockIdx.x * 8 + wave;
  size_t base = (size_t)row * D_MODEL;
  float v[4], s = 0.0f;
#pragma unroll
  for (int i = 0; i < 4; ++i) {
    int c = lane + i * 32;
    v[i] = x[base + c] + o[base + c];
    s += v[i];
  }
#pragma unroll
  for (int m = 1; m < 32; m <<= 1) s += __shfl_xor(s, m, 32);
  float mean = s * (1.0f / 128.0f);
  float var = 0.0f;
#pragma unroll
  for (int i = 0; i < 4; ++i) { float d = v[i] - mean; var += d * d; }
#pragma unroll
  for (int m = 1; m < 32; m <<= 1) var += __shfl_xor(var, m, 32);
  float rs = rsqrtf(var * (1.0f / 128.0f) + 1e-5f);
#pragma unroll
  for (int i = 0; i < 4; ++i) {
    int c = lane + i * 32;
    float y = (v[i] - mean) * rs * g[c] + bb[c];
    x[base + c] = y;
    x16[base + c] = (_Float16)y;
  }
}

// ---------------------------------------------------------------------------
// h_temporal[b,d] = mean over S of x[b,s,d]
// ---------------------------------------------------------------------------
__global__ __launch_bounds__(256) void pool_kernel(
    const float* __restrict__ x, float* __restrict__ ht)
{
  int idx = blockIdx.x * 256 + threadIdx.x;     // 0..2047
  int b = idx >> 7, d = idx & 127;
  float s = 0.0f;
  for (int t = 0; t < SEQ; ++t)
    s += x[(size_t)(b * SEQ + t) * D_MODEL + d];
  ht[idx] = s * (1.0f / (float)SEQ);
}

// ---------------------------------------------------------------------------
// 24-bin hour histogram, dist_feat[b,25] = [dist, cdi], cdi = JS(dist, base)
// ---------------------------------------------------------------------------
__global__ __launch_bounds__(64) void hist_kernel(
    const float* __restrict__ hours, const float* __restrict__ baseline,
    float* __restrict__ df, float* __restrict__ cdi_out)
{
  __shared__ int cnt[24];
  int b = blockIdx.x, t = threadIdx.x;
  if (t < 24) cnt[t] = 0;
  __syncthreads();
  for (int s = t; s < SEQ; s += 64) {
    int bin = (int)hours[b * SEQ + s];
    bin = bin < 0 ? 0 : (bin > 23 ? 23 : bin);
    atomicAdd(&cnt[bin], 1);
  }
  __syncthreads();
  if (t < 24)
    df[b * 25 + t] = (float)cnt[t] / ((float)SEQ + 1e-10f);
  if (t == 0) {
    float p[24], q[24], psum = 0.0f, qsum = 0.0f;
    for (int j = 0; j < 24; ++j) {
      p[j] = (float)cnt[j] / ((float)SEQ + 1e-10f) + 1e-10f;
      q[j] = baseline[b * 24 + j] + 1e-10f;
      psum += p[j]; qsum += q[j];
    }
    float acc = 0.0f;
    for (int j = 0; j < 24; ++j) {
      float pp = p[j] / psum, qq = q[j] / qsum;
      float mm = 0.5f * (pp + qq);
      acc += 0.5f * pp * logf(pp / mm) + 0.5f * qq * logf(qq / mm);
    }
    df[b * 25 + 24] = acc;
    cdi_out[b] = acc;
  }
}

// ---------------------------------------------------------------------------
// Heads: h_dist = df @ w_dist^T + b_dist ; h_cr = [ht|h_dist] @ w_proj^T + b
// ---------------------------------------------------------------------------
__global__ __launch_bounds__(128) void heads_kernel(
    const float* __restrict__ ht, const float* __restrict__ df,
    const float* __restrict__ w_dist, const float* __restrict__ b_dist,
    const float* __restrict__ w_proj, const float* __restrict__ b_proj,
    float* __restrict__ out)
{
  __shared__ float cat[192];
  int b = blockIdx.x, t = threadIdx.x;
  cat[t] = ht[b * 128 + t];
  if (t < 64) {
    float s = b_dist[t];
    for (int j = 0; j < 25; ++j) s += df[b * 25 + j] * w_dist[t * 25 + j];
    cat[128 + t] = s;
  }
  __syncthreads();
  float s = b_proj[t];
  for (int j = 0; j < 192; ++j) s += cat[j] * w_proj[t * 192 + j];
  out[b * 128 + t] = s;
}

// ---------------------------------------------------------------------------
// Launch. Param leaves follow JAX pytree flattening (dict keys sorted):
//  0 activity_ids  1 hours  2 day_of_week  3 baseline
//  4 act_emb  5 b_dist  6 b_proj  7 day_emb
//  8+12*li: b1,b2,b_in,b_out,ln1_b,ln1_g,ln2_b,ln2_g,w1,w2,w_in,w_out
//  44 w_dist  45 w_proj
// ---------------------------------------------------------------------------
extern "C" void kernel_launch(void* const* d_in, const int* in_sizes, int n_in,
                              void* d_out, int out_size, void* d_ws, size_t ws_size,
                              hipStream_t stream)
{
  (void)in_sizes; (void)n_in; (void)out_size; (void)ws_size;
  const int*   act      = (const int*)d_in[0];
  const float* hours    = (const float*)d_in[1];
  const int*   dow      = (const int*)d_in[2];
  const float* baseline = (const float*)d_in[3];
  const float* act_emb  = (const float*)d_in[4];
  const float* b_dist   = (const float*)d_in[5];
  const float* b_proj   = (const float*)d_in[6];
  const float* day_emb  = (const float*)d_in[7];
  const float* w_dist   = (const float*)d_in[44];
  const float* w_proj   = (const float*)d_in[45];

  char* ws = (char*)d_ws;
  size_t off = 0;
  float*    x      = (float*)(ws + off);    off += (size_t)ROWS * 128 * 4;  // 8 MB
  _Float16* x16    = (_Float16*)(ws + off); off += (size_t)ROWS * 128 * 2;  // 4 MB
  _Float16* qkv16  = (_Float16*)(ws + off); off += (size_t)ROWS * 384 * 2;  // 12 MB
  _Float16* ctx16  = (_Float16*)(ws + off); off += (size_t)ROWS * 128 * 2;  // 4 MB
  _Float16* f16b   = (_Float16*)(ws + off); off += (size_t)ROWS * 512 * 2;  // 16 MB
  float*    o32    = (float*)(ws + off);    off += (size_t)ROWS * 128 * 4;  // 8 MB
  _Float16* w_in16  = (_Float16*)(ws + off); off += 384 * 128 * 2;
  _Float16* w_out16 = (_Float16*)(ws + off); off += 128 * 128 * 2;
  _Float16* w1_16   = (_Float16*)(ws + off); off += 512 * 128 * 2;
  _Float16* w2_16   = (_Float16*)(ws + off); off += 128 * 512 * 2;
  float*    ht      = (float*)(ws + off);    off += BATCH * D_MODEL * 4;
  float*    df      = (float*)(ws + off);

  float* out_hcr = (float*)d_out;
  float* out_cdi = out_hcr + BATCH * D_MODEL;

  embed_kernel<<<ROWS * 64 / 256, 256, 0, stream>>>(act, hours, dow, act_emb,
                                                    day_emb, x, x16);

  for (int li = 0; li < 3; ++li) {
    const float* b1    = (const float*)d_in[8 + 12 * li + 0];
    const float* b2    = (const float*)d_in[8 + 12 * li + 1];
    const float* b_in  = (const float*)d_in[8 + 12 * li + 2];
    const float* b_out = (const float*)d_in[8 + 12 * li + 3];
    const float* ln1_b = (const float*)d_in[8 + 12 * li + 4];
    const float* ln1_g = (const float*)d_in[8 + 12 * li + 5];
    const float* ln2_b = (const float*)d_in[8 + 12 * li + 6];
    const float* ln2_g = (const float*)d_in[8 + 12 * li + 7];
    const float* w1    = (const float*)d_in[8 + 12 * li + 8];
    const float* w2    = (const float*)d_in[8 + 12 * li + 9];
    const float* w_in  = (const float*)d_in[8 + 12 * li + 10];
    const float* w_out = (const float*)d_in[8 + 12 * li + 11];

    // convert this layer's weights to f16 once
    cvt_f16_kernel<<<384 * 128 / 1024, 256, 0, stream>>>(w_in,  w_in16);
    cvt_f16_kernel<<<128 * 128 / 1024, 256, 0, stream>>>(w_out, w_out16);
    cvt_f16_kernel<<<512 * 128 / 1024, 256, 0, stream>>>(w1,    w1_16);
    cvt_f16_kernel<<<128 * 512 / 1024, 256, 0, stream>>>(w2,    w2_16);

    // qkv16 = f16(x @ w_in^T + b_in)      [16384, 384]
    gemm16_kernel<<<(ROWS / 16) * (384 / 16) / 8, 256, 0, stream>>>(
        x16, w_in16, b_in, nullptr, qkv16, ROWS, 384, 128, 0);
    // attention -> ctx16                  [16384, 128]
    attn_kernel<<<BATCH * NHEAD * (SEQ / 16) / 8, 256, 0, stream>>>(qkv16, ctx16);
    // o32 = ctx @ w_out^T + b_out         (fp32, for residual)
    gemm16_kernel<<<(ROWS / 16) * (128 / 16) / 8, 256, 0, stream>>>(
        ctx16, w_out16, b_out, o32, nullptr, ROWS, 128, 128, 0);
    // x = LN(x + o); also refresh x16
    add_ln_kernel<<<ROWS / 8, 256, 0, stream>>>(x, o32, ln1_g, ln1_b, x16);
    // f16b = f16(relu(x @ w1^T + b1))     [16384, 512]
    gemm16_kernel<<<(ROWS / 16) * (512 / 16) / 8, 256, 0, stream>>>(
        x16, w1_16, b1, nullptr, f16b, ROWS, 512, 128, 1);
    // o32 = f @ w2^T + b2
    gemm16_kernel<<<(ROWS / 16) * (128 / 16) / 8, 256, 0, stream>>>(
        f16b, w2_16, b2, o32, nullptr, ROWS, 128, 512, 0);
    // x = LN(x + f2); refresh x16
    add_ln_kernel<<<ROWS / 8, 256, 0, stream>>>(x, o32, ln2_g, ln2_b, x16);
  }

  pool_kernel<<<BATCH * D_MODEL / 256, 256, 0, stream>>>(x, ht);
  hist_kernel<<<BATCH, 64, 0, stream>>>(hours, baseline, df, out_cdi);
  heads_kernel<<<BATCH, 128, 0, stream>>>(ht, df, w_dist, b_dist, w_proj, b_proj,
                                          out_hcr);
}
